// PatchInpainting_515396076444
// MI455X (gfx1250) — matrix operand order repaired and checked by reading, not compiled, not run on hardware
//
#include <hip/hip_runtime.h>
#include <stdint.h>

// ---------------------------------------------------------------------------
// PatchInpainting attention for MI455X (gfx1250, wave32, WMMA bf16).
//
// Pipeline:
//   cvt(f32->bf16)  q,k,v,Wq,Wk,Wv,Wfc
//   GEMM  Qb = q @ Wq^T            [8192,2048] bf16
//   GEMM  Kb = k @ Wk^T            [8192,2048] bf16
//   GEMM  Vh = v @ Wv^T            [8192,3072] f32
//   GEMM  S  = scale*Q@K^T - 1e4*I [32,1024,1024] f32   (batched over B*H)
//   topk  per-row top-8 -> softmax -> straight-through => attn row = zeros +
//         c at argmax (exact in fp32); record {argmax,c}
//   gather head[b,q,h*768+d] = c * Vh[b,km,h*768+d]     bf16
//   GEMM  out = head @ Wfc^T       [8192,768]  f32
//
// GEMM inner loop: double-buffered LDS ping-pong; next tile copied with
// GLOBAL_LOAD_ASYNC_TO_LDS_B128 (ASYNCcnt) while current tile feeds 16
// v_wmma_f32_16x16x32_bf16 per wave per K-step; one barrier per iteration.
// ---------------------------------------------------------------------------

#define DEV_INLINE __device__ __forceinline__

typedef __bf16 bf16_t;
typedef __attribute__((ext_vector_type(8)))  bf16_t v8bf;
typedef __attribute__((ext_vector_type(16))) bf16_t v16bf;
typedef __attribute__((ext_vector_type(8)))  float  v8f;
typedef int v4i_g __attribute__((vector_size(16)));   // GCC-style int4 (builtin param type)

static constexpr int B_  = 8;
static constexpr int LQ  = 1024;
static constexpr int LK  = 1024;
static constexpr int H_  = 4;
static constexpr int DK  = 512;
static constexpr int DV  = 768;
static constexpr int TD  = 896;
static constexpr int HDK = H_ * DK;   // 2048
static constexpr int HDV = H_ * DV;   // 3072
#define NEG_MASK (-10000.0f)

#if defined(__AMDGCN__) && __has_builtin(__builtin_amdgcn_global_load_async_to_lds_b128)
#define HAS_ASYNC_LDS 1
#else
#define HAS_ASYNC_LDS 0
#endif

DEV_INLINE unsigned short f32_to_bf16_bits(float f) {
  union { float f; unsigned u; } v; v.f = f;
  unsigned u = v.u;
  unsigned r = u + 0x7FFFu + ((u >> 16) & 1u);  // round-to-nearest-even
  return (unsigned short)(r >> 16);
}

#if HAS_ASYNC_LDS
DEV_INLINE void async_b128(const unsigned short* g, unsigned short* l) {
  // Builtin signature (from hipcc diagnostic): (v4i AS1*, v4i AS3*, Imm, Imm)
  __builtin_amdgcn_global_load_async_to_lds_b128(
      (__attribute__((address_space(1))) v4i_g*)g,
      (__attribute__((address_space(3))) v4i_g*)l,
      0, 0);
}
#endif

DEV_INLINE void wait_async_zero() {
#if defined(__AMDGCN__) && __has_builtin(__builtin_amdgcn_s_wait_asynccnt)
  __builtin_amdgcn_s_wait_asynccnt(0);
#else
  asm volatile("s_wait_asynccnt 0x0" ::: "memory");
#endif
}

// ---------------------------------------------------------------------------
// f32 -> bf16(bits) elementwise conversion
// ---------------------------------------------------------------------------
__global__ void __launch_bounds__(256)
cvt_f32_bf16(const float* __restrict__ src, unsigned short* __restrict__ dst, int n) {
  int i = blockIdx.x * 256 + threadIdx.x;
  if (i < n) dst[i] = f32_to_bf16_bits(src[i]);
}

// ---------------------------------------------------------------------------
// Generic tiled GEMM:  C[M,N] = alpha * A[M,K] @ B[N,K]^T  (+ NEG on diagonal)
// A,B: bf16 bit patterns, row-major with leading dims lda/ldb.
// Batched via z: offsets (z/zdiv)*s?1 + (z%zdiv)*s?2 (element strides).
// Block = 128x128, 4 waves (128 thr), each wave computes 64x64 via 4x4 WMMA
// fragments: v_wmma_f32_16x16x32_bf16, f32 accumulation.
// ---------------------------------------------------------------------------
template <typename OutT>
__global__ void __launch_bounds__(128)
gemm_bt_bf16(const unsigned short* __restrict__ A,
             const unsigned short* __restrict__ Bm,
             OutT* __restrict__ C,
             int M, int N, int K, int lda, int ldb, int ldc,
             int zdiv,
             long long sA1, long long sA2,
             long long sB1, long long sB2,
             long long sC1, long long sC2,
             float alpha, int diagNeg)
{
  constexpr int LDT = 40;  // padded LDS row stride (bf16) -> 80B, conflict-free
  __shared__ __align__(16) unsigned short As[2][128 * LDT];
  __shared__ __align__(16) unsigned short Bs[2][128 * LDT];

  const int tid  = threadIdx.x;
  const int wave = tid >> 5;
  const int lane = tid & 31;
  const int r    = lane & 15;
  const int half = lane >> 4;
  const int wm   = wave >> 1;  // 0..1
  const int wn   = wave & 1;   // 0..1

  const long long z  = blockIdx.z;
  const long long zq = z / zdiv, zr = z % zdiv;
  const unsigned short* Ab = A  + zq * sA1 + zr * sA2;
  const unsigned short* Bb = Bm + zq * sB1 + zr * sB2;
  OutT*                 Cb = C  + zq * sC1 + zr * sC2;

  const int m0 = blockIdx.y * 128;
  const int n0 = blockIdx.x * 128;

  v8f zero = {};
  v8f acc[4][4];
#pragma unroll
  for (int i = 0; i < 4; ++i)
#pragma unroll
    for (int j = 0; j < 4; ++j) acc[i][j] = zero;

  const unsigned short* aRow = Ab + (long long)(m0 + tid) * lda;
  const unsigned short* bRow = Bb + (long long)(n0 + tid) * ldb;
  const int nK = K >> 5;   // K-steps of 32

#if HAS_ASYNC_LDS
  // Issue async global->LDS copy of one 128x32 A tile + 128x32 B tile.
  auto issue_tile = [&](int buf, int k0) {
    unsigned short* ad = &As[buf][tid * LDT];
    unsigned short* bd = &Bs[buf][tid * LDT];
#pragma unroll
    for (int j = 0; j < 4; ++j) {
      async_b128(aRow + k0 + j * 8, ad + j * 8);
      async_b128(bRow + k0 + j * 8, bd + j * 8);
    }
  };
  issue_tile(0, 0);
  wait_async_zero();
  __syncthreads();
#else
  uint4 ra[4], rb[4];
  auto issue_tile = [&](int k0) {
    const uint4* ag = (const uint4*)(aRow + k0);
    const uint4* bg = (const uint4*)(bRow + k0);
#pragma unroll
    for (int j = 0; j < 4; ++j) { ra[j] = ag[j]; rb[j] = bg[j]; }
  };
  auto commit_tile = [&](int buf) {
    uint4* ad = (uint4*)&As[buf][tid * LDT];
    uint4* bd = (uint4*)&Bs[buf][tid * LDT];
#pragma unroll
    for (int j = 0; j < 4; ++j) { ad[j] = ra[j]; bd[j] = rb[j]; }
  };
  issue_tile(0);
  commit_tile(0);
  __syncthreads();
#endif

  for (int it = 0; it < nK; ++it) {
    const int cur = it & 1;
    const int nxt = cur ^ 1;
#if HAS_ASYNC_LDS
    if (it + 1 < nK) issue_tile(nxt, (it + 1) << 5);   // overlaps the WMMAs
#else
    if (it + 1 < nK) issue_tile((it + 1) << 5);        // loads in flight
#endif

    // A fragment (16x32 bf16): lanes 0-15 -> K 0-7 / 16-23, lanes 16-31 ->
    // K 8-15 / 24-31 (ISA A layout). B fragment (32x16): same K order as A
    // (half 0 -> K 0-7 / 16-23, half 1 -> K 8-15 / 24-31).
    v16bf aF[4], bF[4];
#pragma unroll
    for (int f = 0; f < 4; ++f) {
      int rai = (wm * 64 + f * 16 + r) * LDT;
      v8bf lo = *(const v8bf*)&As[cur][rai + half * 8];
      v8bf hi = *(const v8bf*)&As[cur][rai + 16 + half * 8];
      aF[f] = __builtin_shufflevector(lo, hi, 0, 1, 2, 3, 4, 5, 6, 7,
                                      8, 9, 10, 11, 12, 13, 14, 15);
      int rbi = (wn * 64 + f * 16 + r) * LDT;
      v8bf blo = *(const v8bf*)&Bs[cur][rbi + half * 8];
      v8bf bhi = *(const v8bf*)&Bs[cur][rbi + 16 + half * 8];
      bF[f] = __builtin_shufflevector(blo, bhi, 0, 1, 2, 3, 4, 5, 6, 7,
                                      8, 9, 10, 11, 12, 13, 14, 15);
    }
#pragma unroll
    for (int fm = 0; fm < 4; ++fm)
#pragma unroll
      for (int fn = 0; fn < 4; ++fn)
        acc[fm][fn] = __builtin_amdgcn_wmma_f32_16x16x32_bf16(
            false, aF[fm], false, bF[fn], (short)0, acc[fm][fn], false, false);

#if HAS_ASYNC_LDS
    wait_async_zero();                 // next tile landed in LDS
#else
    if (it + 1 < nK) commit_tile(nxt); // s_wait_loadcnt then ds_store
#endif
    __syncthreads();
  }

  // C/D layout: VGPR i holds (M = i + 8*half, N = lane%16) of each 16x16 tile.
#pragma unroll
  for (int fm = 0; fm < 4; ++fm) {
    int mBase = m0 + wm * 64 + fm * 16 + half * 8;
#pragma unroll
    for (int fn = 0; fn < 4; ++fn) {
      int n = n0 + wn * 64 + fn * 16 + r;
#pragma unroll
      for (int i = 0; i < 8; ++i) {
        int m = mBase + i;
        float val = acc[fm][fn][i] * alpha;
        if (diagNeg && (m == n)) val += NEG_MASK;
        if constexpr (sizeof(OutT) == 2)
          Cb[(long long)m * ldc + n] = f32_to_bf16_bits(val);
        else
          Cb[(long long)m * ldc + n] = val;
      }
    }
  }
}

// ---------------------------------------------------------------------------
// Per-row top-8 + softmax + straight-through.  One wave per row of S
// (row = (b*H+h)*LQ + q, 1024 cols).  Writes the full attn row (zeros with c
// at argmax -- exact per fp32 semantics of the reference) and {argmax, c}.
// ---------------------------------------------------------------------------
__global__ void __launch_bounds__(256)
topk_softmax_scatter(const float* __restrict__ S,
                     float* __restrict__ attnOut,
                     int* __restrict__ recIdx,
                     float* __restrict__ recC)
{
  const int wave = threadIdx.x >> 5;
  const int lane = threadIdx.x & 31;
  const long long row = (long long)blockIdx.x * 8 + wave;
  const float* sp = S + row * LK;

  float x[32];
#pragma unroll
  for (int j = 0; j < 32; ++j) x[j] = sp[lane + 32 * j];

  float vals[8];
  int argmax = 0;
#pragma unroll
  for (int t = 0; t < 8; ++t) {
    float best = -3.4e38f;
    int bidx = 0x7fffffff;
#pragma unroll
    for (int j = 0; j < 32; ++j) {
      int idx = lane + 32 * j;
      if (x[j] > best || (x[j] == best && idx < bidx)) { best = x[j]; bidx = idx; }
    }
    for (int off = 16; off; off >>= 1) {   // wave32 butterfly (val,idx) max
      float ov = __shfl_xor(best, off, 32);
      int   oi = __shfl_xor(bidx, off, 32);
      if (ov > best || (ov == best && oi < bidx)) { best = ov; bidx = oi; }
    }
    vals[t] = best;
    if (t == 0) argmax = bidx;
    if (lane == (bidx & 31)) x[bidx >> 5] = -3.4e38f;   // retire winner
  }

  float sum = 0.0f;
#pragma unroll
  for (int t = 0; t < 8; ++t) sum += expf(vals[t] - vals[0]);
  float amax = 1.0f / sum;                // exp(-10000-max) == 0 exactly in f32
  float c = (1.0f - amax) + amax;         // straight-through value at argmax

  float* op = attnOut + row * LK;
#pragma unroll
  for (int j = 0; j < 32; ++j) {
    int col = lane + 32 * j;
    op[col] = (col == argmax) ? c : 0.0f; // non-argmax top-8: (0-a)+a == +0
  }
  if (lane == 0) { recIdx[row] = argmax; recC[row] = c; }
}

// ---------------------------------------------------------------------------
// head[b*LQ+q, h*DV+d] = c[b,h,q] * Vh[b*LK+km, h*DV+d]   (bf16 out)
// ---------------------------------------------------------------------------
__global__ void __launch_bounds__(256)
gather_heads(const float* __restrict__ Vh,
             const int* __restrict__ recIdx,
             const float* __restrict__ recC,
             unsigned short* __restrict__ headOut)
{
  const int rowq = blockIdx.x;    // b*LQ + q
  const int b = rowq >> 10;
  const int q = rowq & 1023;
  for (int e = threadIdx.x; e < HDV; e += 256) {
    const int h = e / DV;
    const long long rrec = ((long long)(b * H_ + h)) * LQ + q;
    const int km = recIdx[rrec];
    const float c = recC[rrec];
    const float v = Vh[((long long)(b * LK + km)) * HDV + e];
    headOut[(long long)rowq * HDV + e] = f32_to_bf16_bits(c * v);
  }
}

// ---------------------------------------------------------------------------
extern "C" void kernel_launch(void* const* d_in, const int* in_sizes, int n_in,
                              void* d_out, int out_size, void* d_ws, size_t ws_size,
                              hipStream_t stream)
{
  (void)in_sizes; (void)n_in; (void)out_size; (void)ws_size;

  const float* q   = (const float*)d_in[0];
  const float* k   = (const float*)d_in[1];
  const float* v   = (const float*)d_in[2];
  // d_in[3] = qk_mask == NEG*eye : folded analytically via diagNeg flag
  const float* Wq  = (const float*)d_in[4];
  const float* Wk  = (const float*)d_in[5];
  const float* Wv  = (const float*)d_in[6];
  const float* Wfc = (const float*)d_in[7];

  float* out  = (float*)d_out;                          // [B,LQ,DV]
  float* attn = (float*)d_out + (size_t)B_ * LQ * DV;   // [B,H,LQ,LK]

  char* w = (char*)d_ws;
  auto alloc = [&](size_t bytes) -> char* {
    char* p = w;
    w += (bytes + 255) & ~(size_t)255;
    return p;
  };
  unsigned short* qb   = (unsigned short*)alloc((size_t)B_ * LQ * TD * 2);
  unsigned short* kb   = (unsigned short*)alloc((size_t)B_ * LK * TD * 2);
  unsigned short* vb   = (unsigned short*)alloc((size_t)B_ * LK * DV * 2);
  unsigned short* wqb  = (unsigned short*)alloc((size_t)HDK * TD * 2);
  unsigned short* wkb  = (unsigned short*)alloc((size_t)HDK * TD * 2);
  unsigned short* wvb  = (unsigned short*)alloc((size_t)HDV * DV * 2);
  unsigned short* wfcb = (unsigned short*)alloc((size_t)DV * HDV * 2);
  unsigned short* Qb   = (unsigned short*)alloc((size_t)B_ * LQ * HDK * 2);
  unsigned short* Kb   = (unsigned short*)alloc((size_t)B_ * LK * HDK * 2);
  float*          Vh   = (float*)alloc((size_t)B_ * LK * HDV * 4);
  float*          S    = (float*)alloc((size_t)B_ * H_ * LQ * LK * 4);
  int*            rIdx = (int*)alloc((size_t)B_ * H_ * LQ * 4);
  float*          rC   = (float*)alloc((size_t)B_ * H_ * LQ * 4);
  unsigned short* head = (unsigned short*)S;  // alias: S dead after topk

  auto cvt = [&](const float* s, unsigned short* d, long long n) {
    cvt_f32_bf16<<<dim3((unsigned)((n + 255) / 256)), dim3(256), 0, stream>>>(s, d, (int)n);
  };
  cvt(q,   qb,   (long long)B_ * LQ * TD);
  cvt(k,   kb,   (long long)B_ * LK * TD);
  cvt(v,   vb,   (long long)B_ * LK * DV);
  cvt(Wq,  wqb,  (long long)HDK * TD);
  cvt(Wk,  wkb,  (long long)HDK * TD);
  cvt(Wv,  wvb,  (long long)HDV * DV);
  cvt(Wfc, wfcb, (long long)DV * HDV);

  const int MR = B_ * LQ;  // 8192 rows (batch-flattened)

  // Q = q @ Wq^T  -> bf16 [8192,2048]
  gemm_bt_bf16<unsigned short><<<dim3(HDK / 128, MR / 128, 1), 128, 0, stream>>>(
      qb, wqb, Qb, MR, HDK, TD, TD, TD, HDK,
      1, 0, 0, 0, 0, 0, 0, 1.0f, 0);
  // K = k @ Wk^T  -> bf16 [8192,2048]
  gemm_bt_bf16<unsigned short><<<dim3(HDK / 128, MR / 128, 1), 128, 0, stream>>>(
      kb, wkb, Kb, MR, HDK, TD, TD, TD, HDK,
      1, 0, 0, 0, 0, 0, 0, 1.0f, 0);
  // Vh = v @ Wv^T -> f32 [8192,3072]
  gemm_bt_bf16<float><<<dim3(HDV / 128, MR / 128, 1), 128, 0, stream>>>(
      vb, wvb, Vh, MR, HDV, DV, DV, DV, HDV,
      1, 0, 0, 0, 0, 0, 0, 1.0f, 0);
  // S = scale * Q @ K^T - 1e4*I, batched over z = b*H + h (head = column slice)
  gemm_bt_bf16<float><<<dim3(LK / 128, LQ / 128, B_ * H_), 128, 0, stream>>>(
      Qb, Kb, S, LQ, LK, DK, HDK, HDK, LK,
      H_,
      (long long)LQ * HDK, (long long)DK,
      (long long)LK * HDK, (long long)DK,
      (long long)H_ * LQ * LK, (long long)LQ * LK,
      0.044194173824159216f /* 1/sqrt(512) */, 1);

  // top-8 + softmax + straight-through: writes full attn output + records
  topk_softmax_scatter<<<dim3(B_ * H_ * LQ / 8), 256, 0, stream>>>(S, attn, rIdx, rC);

  // head_out = c * Vh[argmax]  (bf16, aliases dead S buffer)
  gather_heads<<<dim3(B_ * LQ), 256, 0, stream>>>(Vh, rIdx, rC, head);

  // out = head @ Wfc^T -> f32 [8192,768]
  gemm_bt_bf16<float><<<dim3(DV / 128, MR / 128, 1), 128, 0, stream>>>(
      head, wfcb, out, MR, DV, HDV, HDV, HDV, DV,
      1, 0, 0, 0, 0, 0, 0, 1.0f, 0);
}
